// WhirlpoolV42_36575941493349
// MI455X (gfx1250) — matrix operand order repaired and checked by reading, not compiled, not run on hardware
//
#include <hip/hip_runtime.h>
#include <math.h>
#include <stdint.h>

// ---------------- problem constants (B=1) ----------------
#define T_SEQ  2048
#define DMODEL 1024
#define NH     16
#define NKV    4
#define HD     64
#define QKV_N  1536   // 1024 q + 256 k + 256 v

typedef __attribute__((ext_vector_type(8)))  __bf16 bf16x8;
typedef __attribute__((ext_vector_type(16))) __bf16 bf16x16;
typedef __attribute__((ext_vector_type(8)))  float  f32x8;
typedef __attribute__((ext_vector_type(4)))  unsigned int uint32x4;
typedef __attribute__((ext_vector_type(8)))  int    int32x8;
typedef __attribute__((ext_vector_type(4)))  int    int32x4;

#define DEVI static __device__ __forceinline__

DEVI float bf2f(__bf16 b) {
  unsigned short s = __builtin_bit_cast(unsigned short, b);
  unsigned u = ((unsigned)s) << 16;
  return __builtin_bit_cast(float, u);
}
DEVI __bf16 f2bf(float f) {
  unsigned u = __builtin_bit_cast(unsigned, f);
  unsigned r = u + 0x7FFFu + ((u >> 16) & 1u);   // round-to-nearest-even
  unsigned short s = (unsigned short)(r >> 16);
  return __builtin_bit_cast(__bf16, s);
}
DEVI bf16x16 cat8(bf16x8 a, bf16x8 b) {
  return __builtin_shufflevector(a, b, 0,1,2,3,4,5,6,7,8,9,10,11,12,13,14,15);
}
DEVI f32x8 wmma_bf16(bf16x16 a, bf16x16 b, f32x8 c) {
  // D(f32 16x16) = A(16x32 bf16) * B(32x16 bf16) + C
  return __builtin_amdgcn_wmma_f32_16x16x32_bf16(false, a, false, b, (short)0, c, false, false);
}

// ---- Tensor Data Mover: DMA a 2-D bf16 tile [n_rows x k_cols] (row stride = tensor_k
// elements) from global memory into LDS at byte offset lds_byte_off. D# packed per
// CDNA5 ISA (group0 bits: count/lds_addr/global_addr/type=2;
// group1 bits: data_size=2B, tensor_dim0/1, tile_dim0/1, tensor_dim0_stride).
// This toolchain exposes the 6-arg builtin (g0,g1,g2,g3,g4,cpol). ----
DEVI void tdm_load_2d_bf16(const __bf16* gptr, unsigned lds_byte_off,
                           unsigned n_rows, unsigned k_cols,
                           unsigned tensor_n, unsigned tensor_k) {
  unsigned long long ga = (unsigned long long)(uintptr_t)gptr;
  uint32x4 g0;
  g0[0] = 1u;                                               // count=1 (valid user D#)
  g0[1] = lds_byte_off;                                     // lds_addr (bytes)
  g0[2] = (unsigned)(ga & 0xFFFFFFFFu);                     // global_addr[31:0]
  g0[3] = (unsigned)((ga >> 32) & 0x01FFFFFFu) | (2u << 30);// global_addr[56:32] | type=2
  int32x8 g1;
  g1[0] = (int)(1u << 16);                                  // data_size=1 -> 2 bytes
  g1[1] = (int)((tensor_k & 0xFFFFu) << 16);                // tensor_dim0[15:0]
  g1[2] = (int)((tensor_k >> 16) | ((tensor_n & 0xFFFFu) << 16)); // dim0 hi | dim1 lo
  g1[3] = (int)((tensor_n >> 16) | (k_cols << 16));         // dim1 hi | tile_dim0
  g1[4] = (int)(n_rows & 0xFFFFu);                          // tile_dim1 (tile_dim2=0)
  g1[5] = (int)tensor_k;                                    // tensor_dim0_stride[31:0]
  g1[6] = 0;                                                // stride hi / dim1_stride lo
  g1[7] = 0;
  int32x4 zz4 = {0, 0, 0, 0};
  int32x8 zz8 = {0, 0, 0, 0, 0, 0, 0, 0};
  __builtin_amdgcn_tensor_load_to_lds(g0, g1, zz4, zz4, zz8, 0);
}

// ---------------- fp32 -> bf16 hi/lo split ----------------
__global__ void hypat_split(const float* __restrict__ src,
                            __bf16* __restrict__ h, __bf16* __restrict__ l, int n) {
  int i = blockIdx.x * blockDim.x + threadIdx.x;
  if (i >= n) return;
  float x = src[i];
  __bf16 hi = f2bf(x);
  h[i] = hi;
  l[i] = f2bf(x - bf2f(hi));
}

// ---------------- V transpose + split:  vT[kvh*64+d][t] ----------------
__global__ void hypat_vprep(const float* __restrict__ qkv,
                            __bf16* __restrict__ vTh, __bf16* __restrict__ vTl) {
  int i = blockIdx.x * blockDim.x + threadIdx.x;
  if (i >= T_SEQ * NKV * HD) return;
  int t = i >> 8;          // 256 = NKV*HD per row
  int r = i & 255;         // kvh*64 + d
  float v = qkv[(size_t)t * QKV_N + (DMODEL + NKV * HD) + r];
  __bf16 hv = f2bf(v);
  size_t o = (size_t)r * T_SEQ + t;
  vTh[o] = hv;
  vTl[o] = f2bf(v - bf2f(hv));
}

__global__ void hypat_zero(float* p) { p[0] = 0.0f; }

__global__ void hypat_finalize(const float* __restrict__ acc, float* __restrict__ out) {
  out[0] = acc[0] * (1.0f / (float)(T_SEQ * NH));
}

// ---------------- NT GEMM: C[M][N] = A[M][K] * B[N][K]^T (bf16 hi/lo, f32 acc) ----------------
// block tile 128x128, 8 waves in 4(m) x 2(n), wave tile 32x64.
// B slab (128 x 32, hi+lo) staged in double-buffered LDS by the Tensor Data Mover;
// the DMA of tile k+1 overlaps the 24 WMMAs of tile k.
__global__ __launch_bounds__(256) void hypat_gemm_nt(
    const __bf16* __restrict__ Ah, const __bf16* __restrict__ Al,
    const __bf16* __restrict__ Bh, const __bf16* __restrict__ Bl,
    float* __restrict__ C, int M, int N, int K) {
  extern __shared__ __align__(16) __bf16 sB[];   // 2 bufs x (4096 hi + 4096 lo) bf16 = 32 KB
  const int lane = threadIdx.x & 31;
  const int wv   = threadIdx.x >> 5;
  const int wm   = wv >> 1;          // 0..3
  const int wn   = wv & 1;           // 0..1
  const int hlf  = lane >> 4;
  const int ln   = lane & 15;
  const int bm   = blockIdx.x * 128;
  const int bn   = blockIdx.y * 128;

  const f32x8 vz = {0.f,0.f,0.f,0.f,0.f,0.f,0.f,0.f};
  f32x8 acc[2][4];
#pragma unroll
  for (int i = 0; i < 2; ++i)
#pragma unroll
    for (int j = 0; j < 4; ++j) acc[i][j] = vz;

  const int rowA = bm + wm * 32 + ln;   // + im*16

  // prologue: DMA first B tile (hi+lo planes) into buffer 0
  if (wv == 0) {
    tdm_load_2d_bf16(Bh + (size_t)bn * K, 0u,    128u, 32u, (unsigned)N, (unsigned)K);
    tdm_load_2d_bf16(Bl + (size_t)bn * K, 8192u, 128u, 32u, (unsigned)N, (unsigned)K);
  }

  for (int k0 = 0; k0 < K; k0 += 32) {
    const int buf = (k0 >> 5) & 1;
    if (wv == 0) __builtin_amdgcn_s_wait_tensorcnt(0);  // current tile resident in LDS
    __syncthreads();                                    // LDS visible; other buffer now free
    if (wv == 0 && k0 + 32 < K) {                       // DMA next tile, overlapped w/ compute
      const unsigned boff = (unsigned)(buf ^ 1) * 16384u;
      tdm_load_2d_bf16(Bh + (size_t)bn * K + k0 + 32, boff,         128u, 32u,
                       (unsigned)N, (unsigned)K);
      tdm_load_2d_bf16(Bl + (size_t)bn * K + k0 + 32, boff + 8192u, 128u, 32u,
                       (unsigned)N, (unsigned)K);
    }

    // A fragments straight from global (per-wave rows)
    bf16x16 ah[2], al[2];
#pragma unroll
    for (int im = 0; im < 2; ++im) {
      const size_t ro = (size_t)(rowA + im * 16) * K + k0 + hlf * 8;
      ah[im] = cat8(*(const bf16x8*)(Ah + ro), *(const bf16x8*)(Ah + ro + 16));
      al[im] = cat8(*(const bf16x8*)(Al + ro), *(const bf16x8*)(Al + ro + 16));
    }
    // B fragments from the LDS slab (row-major [128][32])
    const __bf16* bbase = sB + buf * 8192;
    bf16x16 bh[4], bl[4];
#pragma unroll
    for (int jn = 0; jn < 4; ++jn) {
      const int row = wn * 64 + jn * 16 + ln;
      bh[jn] = cat8(*(const bf16x8*)(bbase + row * 32 + hlf * 8), *(const bf16x8*)(bbase + row * 32 + 16 + hlf * 8));
      bl[jn] = cat8(*(const bf16x8*)(bbase + 4096 + row * 32 + hlf * 8), *(const bf16x8*)(bbase + 4096 + row * 32 + 16 + hlf * 8));
    }
#pragma unroll
    for (int im = 0; im < 2; ++im)
#pragma unroll
      for (int jn = 0; jn < 4; ++jn) {
        acc[im][jn] = wmma_bf16(ah[im], bh[jn], acc[im][jn]);   // hi*hi
        acc[im][jn] = wmma_bf16(ah[im], bl[jn], acc[im][jn]);   // hi*lo
        acc[im][jn] = wmma_bf16(al[im], bh[jn], acc[im][jn]);   // lo*hi
      }
    if (k0 + 32 < K)
      __builtin_prefetch(Ah + (size_t)rowA * K + k0 + 32, 0, 1);
  }

#pragma unroll
  for (int im = 0; im < 2; ++im)
#pragma unroll
    for (int jn = 0; jn < 4; ++jn)
#pragma unroll
      for (int r = 0; r < 8; ++r) {
        const int row = bm + wm * 32 + im * 16 + hlf * 8 + r;
        const int col = bn + wn * 64 + jn * 16 + ln;
        C[(size_t)row * N + col] = acc[im][jn][r];
      }
}

// ---------------- fused RMSNorm + RoPE + q0/k0 + spatial-norm ----------------
// one wave per (t, head) row of 64; lane l holds the rope pair (d=l, d=l+32)
__global__ __launch_bounds__(256) void hypat_normrope(
    const float* __restrict__ qkv, const float* __restrict__ curv,
    __bf16* __restrict__ qrh, __bf16* __restrict__ qrl,
    __bf16* __restrict__ krh, __bf16* __restrict__ krl,
    float* __restrict__ q0, float* __restrict__ k0, float* __restrict__ nacc) {
  const int gw   = (int)((blockIdx.x * blockDim.x + threadIdx.x) >> 5);
  const int lane = threadIdx.x & 31;
  const bool isQ = gw < T_SEQ * NH;
  int row = isQ ? gw : gw - T_SEQ * NH;
  if (!isQ && row >= T_SEQ * NKV) return;
  int t, h, cb;
  if (isQ) { t = row >> 4; h = row & 15; cb = h * HD; }
  else     { t = row >> 2; h = row & 3;  cb = DMODEL + h * HD; }

  const float a = qkv[(size_t)t * QKV_N + cb + lane];
  const float b = qkv[(size_t)t * QKV_N + cb + 32 + lane];
  float ss = a * a + b * b;
#pragma unroll
  for (int m = 16; m > 0; m >>= 1) ss += __shfl_xor(ss, m, 32);
  const float scale = rsqrtf(ss * (1.0f / 64.0f) + 1e-6f);
  const float x1 = a * scale, x2 = b * scale;
  const float invf = __powf(10000.0f, -(float)(2 * lane) * (1.0f / 64.0f));
  float cs, sn;
  __sincosf((float)t * invf, &cs, &sn);
  const float an =  x1 * cs + x2 * sn;
  const float bn = -x1 * sn + x2 * cs;
  float s2 = an * an + bn * bn;
#pragma unroll
  for (int m = 16; m > 0; m >>= 1) s2 += __shfl_xor(s2, m, 32);
  const float c  = curv[0];
  const float zc = sqrtf(1.0f / c + s2);

  if (isQ) {
    const size_t o = (size_t)t * DMODEL + h * HD + lane;
    __bf16 ha = f2bf(an); qrh[o]      = ha; qrl[o]      = f2bf(an - bf2f(ha));
    __bf16 hb = f2bf(bn); qrh[o + 32] = hb; qrl[o + 32] = f2bf(bn - bf2f(hb));
    if (lane == 0) {
      q0[(size_t)h * T_SEQ + t] = zc;
      atomicAdd(nacc, sqrtf(s2));
    }
  } else {
    const size_t o = ((size_t)t * NKV + h) * HD + lane;
    __bf16 ha = f2bf(an); krh[o]      = ha; krl[o]      = f2bf(an - bf2f(ha));
    __bf16 hb = f2bf(bn); krh[o + 32] = hb; krl[o + 32] = f2bf(bn - bf2f(hb));
    if (lane == 0) k0[(size_t)h * T_SEQ + t] = zc;
  }
}

// ---------------- flash-style hyperbolic attention ----------------
// one wave per (head, 16-query tile); key blocks of 32
__global__ __launch_bounds__(32) void hypat_attention(
    const __bf16* __restrict__ qrh, const __bf16* __restrict__ qrl,   // [T][H][64]
    const __bf16* __restrict__ krh, const __bf16* __restrict__ krl,   // [T][HKV][64]
    const __bf16* __restrict__ vTh, const __bf16* __restrict__ vTl,   // [HKV*64][T]
    const float* __restrict__ q0, const float* __restrict__ k0,       // [H][T],[HKV][T]
    const float* __restrict__ tempv, const float* __restrict__ curv,
    __bf16* __restrict__ ath, __bf16* __restrict__ atl)               // [T][1024]
{
  __shared__ __align__(16) __bf16 sP[16 * 32];
  const int qt   = blockIdx.x;
  const int h    = blockIdx.y;
  const int kvh  = h >> 2;
  const int lane = threadIdx.x;
  const int hlf  = lane >> 4, ln = lane & 15;
  const int qr0  = qt * 16;

  const float c    = curv[0];
  const float rsc  = rsqrtf(c);
  const float itmp = 1.0f / tempv[h];

  bf16x16 aqh[2], aql[2];
#pragma unroll
  for (int cc = 0; cc < 2; ++cc) {
    const size_t ro = ((size_t)(qr0 + ln) * NH + h) * HD + cc * 32 + hlf * 8;
    aqh[cc] = cat8(*(const bf16x8*)(qrh + ro), *(const bf16x8*)(qrh + ro + 16));
    aql[cc] = cat8(*(const bf16x8*)(qrl + ro), *(const bf16x8*)(qrl + ro + 16));
  }
  float q0v[8];
#pragma unroll
  for (int r = 0; r < 8; ++r) q0v[r] = q0[(size_t)h * T_SEQ + qr0 + hlf * 8 + r];

  const f32x8 vz = {0.f,0.f,0.f,0.f,0.f,0.f,0.f,0.f};
  f32x8 oacc[4] = {vz, vz, vz, vz};
  float mrun[8], lrun[8];
#pragma unroll
  for (int r = 0; r < 8; ++r) { mrun[r] = -__builtin_inff(); lrun[r] = 0.f; }

  const int nkb = (qr0 + 15) / 32 + 1;
  for (int kb = 0; kb < nkb; ++kb) {
    const int kbase = kb * 32;
    f32x8 s[2] = {vz, vz};
#pragma unroll
    for (int tt = 0; tt < 2; ++tt)
#pragma unroll
      for (int cc = 0; cc < 2; ++cc) {
        const size_t ro = ((size_t)(kbase + tt * 16 + ln) * NKV + kvh) * HD + cc * 32 + hlf * 8;
        bf16x16 kh = cat8(*(const bf16x8*)(krh + ro), *(const bf16x8*)(krh + ro + 16));
        bf16x16 kl = cat8(*(const bf16x8*)(krl + ro), *(const bf16x8*)(krl + ro + 16));
        s[tt] = wmma_bf16(aqh[cc], kh, s[tt]);
        s[tt] = wmma_bf16(aqh[cc], kl, s[tt]);
        s[tt] = wmma_bf16(aql[cc], kh, s[tt]);
      }
    const float k0a = k0[(size_t)kvh * T_SEQ + kbase + ln];
    const float k0b = k0[(size_t)kvh * T_SEQ + kbase + 16 + ln];

    float alpha[8];
#pragma unroll
    for (int r = 0; r < 8; ++r) {
      const int qi = qr0 + hlf * 8 + r;
      float mink0 = s[0][r] - q0v[r] * k0a;
      float xx0   = fmaxf(-c * mink0, 1.0f + 1e-5f);
      float sc0   = ((kbase + ln) <= qi)
                  ? (-(__logf(xx0 + sqrtf(xx0 * xx0 - 1.0f)) * rsc) * itmp)
                  : -__builtin_inff();
      float mink1 = s[1][r] - q0v[r] * k0b;
      float xx1   = fmaxf(-c * mink1, 1.0f + 1e-5f);
      float sc1   = ((kbase + 16 + ln) <= qi)
                  ? (-(__logf(xx1 + sqrtf(xx1 * xx1 - 1.0f)) * rsc) * itmp)
                  : -__builtin_inff();
      float mx = fmaxf(sc0, sc1);
#pragma unroll
      for (int mm = 8; mm > 0; mm >>= 1) mx = fmaxf(mx, __shfl_xor(mx, mm, 32));
      const float mnew = fmaxf(mrun[r], mx);
      const float p0 = __expf(sc0 - mnew);
      const float p1 = __expf(sc1 - mnew);
      sP[(hlf * 8 + r) * 32 + ln]      = f2bf(p0);
      sP[(hlf * 8 + r) * 32 + 16 + ln] = f2bf(p1);
      float rs = p0 + p1;
#pragma unroll
      for (int mm = 8; mm > 0; mm >>= 1) rs += __shfl_xor(rs, mm, 32);
      alpha[r] = __expf(mrun[r] - mnew);
      lrun[r]  = lrun[r] * alpha[r] + rs;
      mrun[r]  = mnew;
    }
#pragma unroll
    for (int dt = 0; dt < 4; ++dt)
#pragma unroll
      for (int r = 0; r < 8; ++r) oacc[dt][r] *= alpha[r];

    __syncthreads();   // C-layout probs in LDS -> re-read in A-layout
    bf16x16 pa = cat8(*(const bf16x8*)(sP + ln * 32 + hlf * 8),
                      *(const bf16x8*)(sP + ln * 32 + 16 + hlf * 8));
    __syncthreads();

#pragma unroll
    for (int dt = 0; dt < 4; ++dt) {
      const size_t ro = ((size_t)(kvh * HD + dt * 16 + ln)) * T_SEQ + kbase + hlf * 8;
      bf16x16 vh = cat8(*(const bf16x8*)(vTh + ro), *(const bf16x8*)(vTh + ro + 16));
      bf16x16 vl = cat8(*(const bf16x8*)(vTl + ro), *(const bf16x8*)(vTl + ro + 16));
      oacc[dt] = wmma_bf16(pa, vh, oacc[dt]);
      oacc[dt] = wmma_bf16(pa, vl, oacc[dt]);
    }
  }

#pragma unroll
  for (int dt = 0; dt < 4; ++dt)
#pragma unroll
    for (int r = 0; r < 8; ++r) {
      const float o = oacc[dt][r] / lrun[r];
      const size_t idx = (size_t)(qr0 + hlf * 8 + r) * DMODEL + h * HD + dt * 16 + ln;
      __bf16 ho = f2bf(o);
      ath[idx] = ho;
      atl[idx] = f2bf(o - bf2f(ho));
    }
}

// ---------------- launcher ----------------
extern "C" void kernel_launch(void* const* d_in, const int* in_sizes, int n_in,
                              void* d_out, int out_size, void* d_ws, size_t ws_size,
                              hipStream_t stream) {
  (void)in_sizes; (void)n_in; (void)out_size; (void)ws_size;
  const float* x    = (const float*)d_in[0];
  const float* q_w  = (const float*)d_in[1];
  const float* k_w  = (const float*)d_in[2];
  const float* v_w  = (const float*)d_in[3];
  const float* o_w  = (const float*)d_in[4];
  const float* temp = (const float*)d_in[5];
  const float* curv = (const float*)d_in[6];
  float* out = (float*)d_out;

  char* ws = (char*)d_ws;
  size_t off = 0;
  auto take = [&](size_t bytes) -> char* {
    char* p = ws + off;
    off = (off + bytes + 255) & ~(size_t)255;
    return p;
  };

  const size_t NX  = (size_t)T_SEQ * DMODEL;        // 2,097,152
  const size_t NWQ = (size_t)DMODEL * DMODEL;       // 1,048,576
  const size_t NWK = (size_t)NKV * HD * DMODEL;     //   262,144
  const size_t NKV_E = (size_t)T_SEQ * NKV * HD;    //   524,288

  __bf16* xh  = (__bf16*)take(NX * 2);
  __bf16* xl  = (__bf16*)take(NX * 2);
  __bf16* wh  = (__bf16*)take((NWQ + 2 * NWK) * 2);
  __bf16* wl  = (__bf16*)take((NWQ + 2 * NWK) * 2);
  __bf16* owh = (__bf16*)take(NWQ * 2);
  __bf16* owl = (__bf16*)take(NWQ * 2);
  float*  qkv = (float*)take((size_t)T_SEQ * QKV_N * 4);
  __bf16* qrh = (__bf16*)take(NX * 2);
  __bf16* qrl = (__bf16*)take(NX * 2);
  __bf16* krh = (__bf16*)take(NKV_E * 2);
  __bf16* krl = (__bf16*)take(NKV_E * 2);
  __bf16* vth = (__bf16*)take(NKV_E * 2);
  __bf16* vtl = (__bf16*)take(NKV_E * 2);
  float*  q0  = (float*)take((size_t)NH * T_SEQ * 4);
  float*  k0  = (float*)take((size_t)NKV * T_SEQ * 4);
  __bf16* ath = (__bf16*)take(NX * 2);
  __bf16* atl = (__bf16*)take(NX * 2);
  float*  nacc = (float*)take(256);

  hypat_zero<<<1, 1, 0, stream>>>(nacc);

  hypat_split<<<(int)((NX  + 255) / 256), 256, 0, stream>>>(x,   xh,  xl,  (int)NX);
  hypat_split<<<(int)((NWQ + 255) / 256), 256, 0, stream>>>(q_w, wh,  wl,  (int)NWQ);
  hypat_split<<<(int)((NWK + 255) / 256), 256, 0, stream>>>(k_w, wh + NWQ,       wl + NWQ,       (int)NWK);
  hypat_split<<<(int)((NWK + 255) / 256), 256, 0, stream>>>(v_w, wh + NWQ + NWK, wl + NWQ + NWK, (int)NWK);
  hypat_split<<<(int)((NWQ + 255) / 256), 256, 0, stream>>>(o_w, owh, owl, (int)NWQ);

  // QKV projection: [2048 x 1536] = x[2048 x 1024] * Wqkv[1536 x 1024]^T
  hypat_gemm_nt<<<dim3(T_SEQ / 128, QKV_N / 128), 256, 32768, stream>>>(
      xh, xl, wh, wl, qkv, T_SEQ, QKV_N, DMODEL);

  hypat_normrope<<<(T_SEQ * NH + T_SEQ * NKV) / 8, 256, 0, stream>>>(
      qkv, curv, qrh, qrl, krh, krl, q0, k0, nacc);

  hypat_vprep<<<(int)(NKV_E / 256), 256, 0, stream>>>(qkv, vth, vtl);

  hypat_attention<<<dim3(T_SEQ / 16, NH), 32, 0, stream>>>(
      qrh, qrl, krh, krl, vth, vtl, q0, k0, temp, curv, ath, atl);

  // output projection: out[2048 x 1024] = attn[2048 x 1024] * o_w[1024 x 1024]^T
  hypat_gemm_nt<<<dim3(T_SEQ / 128, DMODEL / 128), 256, 32768, stream>>>(
      ath, atl, owh, owl, out, T_SEQ, DMODEL, DMODEL);

  hypat_finalize<<<1, 1, 0, stream>>>(nacc, out + NX);
}
